// AMSignQuantize_38036230373922
// MI455X (gfx1250) — 8-bit matrix operand order repaired and checked by reading, compiled, not run on hardware
//
#include <hip/hip_runtime.h>
#include <stdint.h>

#define BATCH 8192
#define DIM   10240
#define NCLS  1000
#define NPAD  1024

#define KSTEP     64
#define LDS_ROW   80
#define TILE_B    (128 * LDS_ROW)
#define BUF_B     (2 * TILE_B)

typedef __attribute__((ext_vector_type(8))) int v8i;

__device__ __forceinline__ uint32_t pack_signs4(float4 v) {
    uint32_t b0 = (v.x > 0.0f) ? 0x01u : 0xFFu;
    uint32_t b1 = (v.y > 0.0f) ? 0x01u : 0xFFu;
    uint32_t b2 = (v.z > 0.0f) ? 0x01u : 0xFFu;
    uint32_t b3 = (v.w > 0.0f) ? 0x01u : 0xFFu;
    return b0 | (b1 << 8) | (b2 << 16) | (b3 << 24);
}

__global__ void hd_quantize_query_kernel(const float* __restrict__ in,
                                         int8_t* __restrict__ out) {
    size_t i = (size_t)blockIdx.x * blockDim.x + threadIdx.x;
    float4 v = ((const float4*)in)[i];
    ((uint32_t*)out)[i] = pack_signs4(v);
}

__global__ void hd_quantize_am_kernel(const float* __restrict__ in,
                                      int8_t* __restrict__ out) {
    size_t i   = (size_t)blockIdx.x * blockDim.x + threadIdx.x;
    size_t row = i / (DIM / 4);
    uint32_t packed = 0u;
    if (row < NCLS) {
        float4 v = ((const float4*)in)[i];
        packed = pack_signs4(v);
    }
    ((uint32_t*)out)[i] = packed;
}

__device__ __forceinline__ void async_copy_b128(const int8_t* g, int8_t* l) {
    unsigned lofs = (unsigned)(size_t)l;
    asm volatile("global_load_async_to_lds_b128 %0, %1, off"
                 :: "v"(lofs), "v"(g) : "memory");
}
__device__ __forceinline__ void wait_async_all() {
    asm volatile("s_wait_asynccnt 0x0" ::: "memory");
}

__global__ void __launch_bounds__(256, 2)
hd_gemm_iu8_lds_kernel(const int8_t* __restrict__ Aq,
                       const int8_t* __restrict__ Bq,
                       float* __restrict__ out) {
    __shared__ __align__(16) int8_t smem[2 * BUF_B];

    const int tid   = threadIdx.x;
    const int lane  = tid & 31;
    const int wave  = tid >> 5;
    const int waveM = wave >> 1;
    const int waveN = wave & 1;
    const int half  = lane >> 4;
    const int l16   = lane & 15;

    const int cA0 = tid, cA1 = tid + 256;
    const int rA0 = cA0 >> 2, oA0 = (cA0 & 3) * 16;
    const int rA1 = cA1 >> 2, oA1 = (cA1 & 3) * 16;

    const int8_t* gA0 = Aq + (size_t)(blockIdx.y * 128 + rA0) * DIM + oA0;
    const int8_t* gA1 = Aq + (size_t)(blockIdx.y * 128 + rA1) * DIM + oA1;
    const int8_t* gB0 = Bq + (size_t)(blockIdx.x * 128 + rA0) * DIM + oA0;
    const int8_t* gB1 = Bq + (size_t)(blockIdx.x * 128 + rA1) * DIM + oA1;

    const int lA0 = rA0 * LDS_ROW + oA0;
    const int lA1 = rA1 * LDS_ROW + oA1;

    auto stage = [&](int buf, int k0) {
        int8_t* base = smem + buf * BUF_B;
        async_copy_b128(gA0 + k0, base + lA0);
        async_copy_b128(gA1 + k0, base + lA1);
        async_copy_b128(gB0 + k0, base + TILE_B + lA0);
        async_copy_b128(gB1 + k0, base + TILE_B + lA1);
    };

    v8i acc[2][4] = {};

    stage(0, 0);

    for (int k0 = 0, it = 0; k0 < DIM; k0 += KSTEP, ++it) {
        wait_async_all();
        __syncthreads();
        if (k0 + KSTEP < DIM) stage((it + 1) & 1, k0 + KSTEP);

        const int8_t* bufA = smem + (it & 1) * BUF_B;
        const int8_t* bufB = bufA + TILE_B;

        v8i afr[2];
#pragma unroll
        for (int mi = 0; mi < 2; ++mi) {
            const int8_t* p = bufA + (waveM * 32 + mi * 16 + l16) * LDS_ROW + half * 16;
            const int2* q = (const int2*)p;
            int2 x0 = q[0], x1 = q[1], x2 = q[4], x3 = q[5];
            afr[mi][0] = x0.x; afr[mi][1] = x0.y;
            afr[mi][2] = x1.x; afr[mi][3] = x1.y;
            afr[mi][4] = x2.x; afr[mi][5] = x2.y;
            afr[mi][6] = x3.x; afr[mi][7] = x3.y;
        }
        v8i bfr[4];
#pragma unroll
        for (int ni = 0; ni < 4; ++ni) {
            const int8_t* p = bufB + (waveN * 64 + ni * 16 + l16) * LDS_ROW + half * 16;
            const int4* q = (const int4*)p;
            int4 x0 = q[0], x1 = q[2];
            bfr[ni][0] = x0.x; bfr[ni][1] = x0.y; bfr[ni][2] = x0.z; bfr[ni][3] = x0.w;
            bfr[ni][4] = x1.x; bfr[ni][5] = x1.y; bfr[ni][6] = x1.z; bfr[ni][7] = x1.w;
        }

#pragma unroll
        for (int mi = 0; mi < 2; ++mi)
#pragma unroll
            for (int ni = 0; ni < 4; ++ni)
                acc[mi][ni] = __builtin_amdgcn_wmma_i32_16x16x64_iu8(
                    true, afr[mi], true, bfr[ni], acc[mi][ni], false, false);
    }

    const int mBase = blockIdx.y * 128 + waveM * 32;
    const int nBase = blockIdx.x * 128 + waveN * 64;
#pragma unroll
    for (int mi = 0; mi < 2; ++mi) {
#pragma unroll
        for (int r = 0; r < 8; ++r) {
            const int row = mBase + mi * 16 + half * 8 + r;
#pragma unroll
            for (int ni = 0; ni < 4; ++ni) {
                const int col = nBase + ni * 16 + l16;
                if (col < NCLS)
                    out[(size_t)row * NCLS + col] = (float)(DIM + acc[mi][ni][r]) * 0.5f;
            }
        }
    }
}

extern "C" void kernel_launch(void* const* d_in, const int* in_sizes, int n_in,
                              void* d_out, int out_size, void* d_ws, size_t ws_size,
                              hipStream_t stream) {
    const float* query = (const float*)d_in[0];
    const float* am    = (const float*)d_in[1];
    float*       out   = (float*)d_out;

    int8_t* Aq = (int8_t*)d_ws;
    int8_t* Bq = Aq + (size_t)BATCH * DIM;

    {
        size_t n4 = (size_t)BATCH * DIM / 4;
        hd_quantize_query_kernel<<<dim3((unsigned)(n4 / 256)), 256, 0, stream>>>(query, Aq);
    }
    {
        size_t n4 = (size_t)NPAD * DIM / 4;
        hd_quantize_am_kernel<<<dim3((unsigned)(n4 / 256)), 256, 0, stream>>>(am, Bq);
    }
    hd_gemm_iu8_lds_kernel<<<dim3(NPAD / 128, BATCH / 128), 256, 0, stream>>>(Aq, Bq, out);
}
